// GCN_4183298146782
// MI455X (gfx1250) — matrix operand order repaired and checked by reading, not compiled, not run on hardware
//
#include <hip/hip_runtime.h>
#include <hip/hip_bf16.h>
#include <stdint.h>

typedef __attribute__((ext_vector_type(16))) __bf16       v16bf;
typedef __attribute__((ext_vector_type(8)))  float        v8f;
typedef __attribute__((ext_vector_type(4)))  unsigned int v4u;
typedef __attribute__((ext_vector_type(8)))  int          v8i;
typedef __attribute__((ext_vector_type(4)))  int          v4i;

#define INDIM 768
#define HID   128

// LDS layout with TDM padding: stream position p (dwords) lands at p + 4*(p/256).
__device__ __forceinline__ int lds_pad_idx(int p) { return p + ((p >> 8) << 2); }

// ---------------- utility fills ----------------
__global__ void k_fill(float* __restrict__ p, float v, int n) {
  int i = blockIdx.x * blockDim.x + threadIdx.x;
  if (i < n) p[i] = v;
}

// ---------------- degree / rsqrt ----------------
__global__ void k_degree(const int* __restrict__ dst, float* __restrict__ deg, int E) {
  int i = blockIdx.x * blockDim.x + threadIdx.x;
  if (i < E) atomicAdd(&deg[dst[i]], 1.0f);
}

__global__ void k_rsqrt(float* __restrict__ d, int n) {
  int i = blockIdx.x * blockDim.x + threadIdx.x;
  if (i < n) d[i] = rsqrtf(d[i]);   // deg >= 1 always (self-loop folded)
}

// ---------------- weight repack into WMMA B-fragment layout (bf16) ----------------
// B fragment (16-bit, 32x16 tile): lane l holds column n = l&15;
// vector element ii maps to k = kstep*32 + ((l>>4)<<3) + (ii&7) + ((ii>>3)<<4).
__global__ void k_repack(const float* __restrict__ W, __bf16* __restrict__ Wf,
                         int K, int Ntiles) {
  const int Ksteps = K >> 5;
  const int total  = Ksteps * Ntiles * 512;   // 32 lanes * 16 elems per (nt,kstep)
  int i = blockIdx.x * blockDim.x + threadIdx.x;
  if (i >= total) return;
  int ii   = i & 15;
  int lane = (i >> 4) & 31;
  int blk  = i >> 9;                // nt*Ksteps + kst
  int kst  = blk % Ksteps;
  int nt   = blk / Ksteps;
  int k = kst * 32 + ((lane >> 4) << 3) + (ii & 7) + ((ii >> 3) << 4);
  int n = nt * 16 + (lane & 15);
  Wf[i] = (__bf16)W[(size_t)k * (Ntiles * 16) + n];
}

// ---------------- bf16 WMMA GEMM with TDM-staged A tile ----------------
// H[M,N] = X[M,K] @ W[K,N].  block = 256 threads = 8 waves; wave w owns N-tile w;
// blockIdx.x = M-tile.  M%16==0, K%32==0, N==128.
// Wave 0 issues one tensor_load_to_lds for the 16xK fp32 A-tile (TDM pads every
// 256 DWORDs by 4 DWORDs -> row stride 780 dwords for K=768: bank-conflict-free,
// 16B aligned). All waves read A-fragments from LDS (ds_load_b128), convert to
// bf16 in registers, and feed v_wmma_f32_16x16x32_bf16.
__global__ void k_gemm_bf16(const float* __restrict__ X, const __bf16* __restrict__ Wf,
                            float* __restrict__ H, int K, int N) {
  extern __shared__ float lds_x[];
  const int Ksteps = K >> 5;
  const int lane = threadIdx.x & 31;
  const int nt   = threadIdx.x >> 5;        // 0..7
  const int mt   = blockIdx.x;
  const int hi16 = lane >> 4;               // lane group 0/1
  const int m    = lane & 15;

#if __has_builtin(__builtin_amdgcn_tensor_load_to_lds)
  if (threadIdx.x < 32) {
    const unsigned long long gaddr =
        (unsigned long long)(uintptr_t)(X + (size_t)mt * 16 * K);
    const unsigned lds_off = (unsigned)(uintptr_t)(void*)lds_x;  // low 32b = LDS offset

    v4u g0;
    g0[0] = 1u;                                    // count=1 (valid descriptor)
    g0[1] = lds_off;                               // lds_addr
    g0[2] = (unsigned)gaddr;                       // global_addr[31:0]
    g0[3] = ((unsigned)(gaddr >> 32) & 0x1FFFFFFu) // global_addr[56:32]
            | (2u << 30);                          // type=2 ("image")

    v8i g1;
    g1[0] = (2 << 16)     // data_size = 4 bytes
          | (1 << 20)     // pad_enable
          | (7 << 22)     // pad_interval code 7 -> every 256 DWORDs
          | (3 << 25);    // pad_amount  code 3 -> 4 DWORDs
    g1[1] = (K & 0xFFFF) << 16;        // tensor_dim0[15:0]   (bits 63:48)
    g1[2] = (16 << 16);                // tensor_dim0[31:16]=0 | tensor_dim1=16
    g1[3] = (K & 0xFFFF) << 16;        // tensor_dim1[31:16]=0 | tile_dim0=K
    g1[4] = 16;                        // tile_dim1=16 | tile_dim2=0
    g1[5] = K;                         // tensor_dim0_stride[31:0]
    g1[6] = 0;                         // stride[47:32]=0 | dim1_stride[15:0]=0
    g1[7] = 0;

    v4i g2 = {0, 0, 0, 0};             // 2D tile: dims 2..4 unused
    v4i g3 = {0, 0, 0, 0};
    v8i g4 = {0, 0, 0, 0, 0, 0, 0, 0}; // extra group (6-arg toolchain form), unused

    __builtin_amdgcn_tensor_load_to_lds(g0, g1, g2, g3, g4, /*cpol=*/0);
    __builtin_amdgcn_s_wait_tensorcnt(0);
  }
#else
  // Fallback: cooperative staged copy into the same padded layout.
  for (int p = threadIdx.x; p < 16 * K; p += blockDim.x)
    lds_x[lds_pad_idx(p)] = X[(size_t)mt * 16 * K + p];
#endif
  __syncthreads();

  const __bf16* wb = Wf + ((size_t)nt * Ksteps * 32 + lane) * 16;

  v8f c = {};
  for (int ks = 0; ks < Ksteps; ++ks) {
    // A fragment (16x32 bf16): lane row m; k = ks*32 + (hi16<<3) + {0..7, 16..23}
    const int kb = (ks << 5) + (hi16 << 3);
    const int p0 = m * K + kb;
    const int p1 = p0 + 16;
    const float* r0 = lds_x + lds_pad_idx(p0);
    const float* r1 = lds_x + lds_pad_idx(p1);
    float4 f0 = *(const float4*)(r0);
    float4 f1 = *(const float4*)(r0 + 4);
    float4 f2 = *(const float4*)(r1);
    float4 f3 = *(const float4*)(r1 + 4);
    v16bf a;
    a[0]  = (__bf16)f0.x; a[1]  = (__bf16)f0.y; a[2]  = (__bf16)f0.z; a[3]  = (__bf16)f0.w;
    a[4]  = (__bf16)f1.x; a[5]  = (__bf16)f1.y; a[6]  = (__bf16)f1.z; a[7]  = (__bf16)f1.w;
    a[8]  = (__bf16)f2.x; a[9]  = (__bf16)f2.y; a[10] = (__bf16)f2.z; a[11] = (__bf16)f2.w;
    a[12] = (__bf16)f3.x; a[13] = (__bf16)f3.y; a[14] = (__bf16)f3.z; a[15] = (__bf16)f3.w;

    v16bf b = *(const v16bf*)(wb + (size_t)ks * 512);

    c = __builtin_amdgcn_wmma_f32_16x16x32_bf16(
        /*neg_a=*/false, a, /*neg_b=*/false, b,
        /*c_mod=*/(short)0, c, /*reuse_a=*/false, /*reuse_b=*/false);
  }

  // C/D layout: vgpr r, lane l -> m = r + ((l>>4)<<3), n = l&15
  const int n = nt * 16 + (lane & 15);
  float* hout = H + (size_t)(mt * 16 + (hi16 << 3)) * N + n;
#pragma unroll
  for (int r = 0; r < 8; ++r) hout[(size_t)r * N] = c[r];
}

// ---------------- edge scatter: acc[dst] += dinv[src] * H[src]  (128 feats) ----------------
// one wave per edge; each lane owns 4 consecutive features.
__global__ void k_scatter128(const float* __restrict__ H, const int* __restrict__ src,
                             const int* __restrict__ dst, const float* __restrict__ dinv,
                             float* __restrict__ acc, int E) {
  const int e = blockIdx.x * (blockDim.x >> 5) + (threadIdx.x >> 5);
  if (e >= E) return;
  const int lane = threadIdx.x & 31;
  const int s = src[e];
  const int d = dst[e];
  const float w = dinv[s];
  const float4 v = *(const float4*)(H + (size_t)s * HID + lane * 4);
  float* out = acc + (size_t)d * HID + lane * 4;
  atomicAdd(out + 0, v.x * w);
  atomicAdd(out + 1, v.y * w);
  atomicAdd(out + 2, v.z * w);
  atomicAdd(out + 3, v.w * w);
}

// ---------------- fused self-loop + norm + bias + (relu) ----------------
// acc = relu( dinv[v] * (acc + dinv[v]*H) + b )   (in place on acc)
__global__ void k_finalize(const float* __restrict__ H, const float* __restrict__ dinv,
                           const float* __restrict__ b, float* __restrict__ acc,
                           int total, int F, int do_relu) {
  int i = blockIdx.x * blockDim.x + threadIdx.x;
  if (i >= total) return;
  int v = i / F, f = i - v * F;
  float dv  = dinv[v];
  float val = dv * (acc[i] + dv * H[i]) + b[f];
  acc[i] = do_relu ? fmaxf(val, 0.0f) : val;
}

// ---------------- classifier linear: hc[v,0..1] = H[v,:] @ Wc ----------------
__global__ void k_linear_c(const float* __restrict__ H, const float* __restrict__ Wc,
                           float* __restrict__ hc, int Nn) {
  int v = blockIdx.x * blockDim.x + threadIdx.x;
  if (v >= Nn) return;
  float a0 = 0.f, a1 = 0.f;
  const float* hr = H + (size_t)v * HID;
#pragma unroll 4
  for (int k = 0; k < HID; k += 4) {
    float4 h4 = *(const float4*)(hr + k);
    a0 += h4.x * Wc[(k + 0) * 2 + 0] + h4.y * Wc[(k + 1) * 2 + 0]
        + h4.z * Wc[(k + 2) * 2 + 0] + h4.w * Wc[(k + 3) * 2 + 0];
    a1 += h4.x * Wc[(k + 0) * 2 + 1] + h4.y * Wc[(k + 1) * 2 + 1]
        + h4.z * Wc[(k + 2) * 2 + 1] + h4.w * Wc[(k + 3) * 2 + 1];
  }
  hc[(size_t)v * 2 + 0] = a0;
  hc[(size_t)v * 2 + 1] = a1;
}

// ---------------- 2-wide edge scatter for classifier layer ----------------
__global__ void k_scatter2(const float* __restrict__ hc, const int* __restrict__ src,
                           const int* __restrict__ dst, const float* __restrict__ dinv,
                           float* __restrict__ acc, int E) {
  int e = blockIdx.x * blockDim.x + threadIdx.x;
  if (e >= E) return;
  int s = src[e], d = dst[e];
  float w = dinv[s];
  atomicAdd(&acc[(size_t)d * 2 + 0], hc[(size_t)s * 2 + 0] * w);
  atomicAdd(&acc[(size_t)d * 2 + 1], hc[(size_t)s * 2 + 1] * w);
}

// ---------------- gather first node of each graph + finalize layer 3 ----------------
__global__ void k_gather(const int* __restrict__ ptr, const float* __restrict__ hc,
                         const float* __restrict__ acc3, const float* __restrict__ dinv,
                         const float* __restrict__ bc, float* __restrict__ out, int G) {
  int i = blockIdx.x * blockDim.x + threadIdx.x;
  if (i >= G) return;
  int p = ptr[i];
  float dv = dinv[p];
  out[i * 2 + 0] = dv * (acc3[(size_t)p * 2 + 0] + dv * hc[(size_t)p * 2 + 0]) + bc[0];
  out[i * 2 + 1] = dv * (acc3[(size_t)p * 2 + 1] + dv * hc[(size_t)p * 2 + 1]) + bc[1];
}

extern "C" void kernel_launch(void* const* d_in, const int* in_sizes, int n_in,
                              void* d_out, int out_size, void* d_ws, size_t ws_size,
                              hipStream_t stream) {
  const float* x  = (const float*)d_in[0];
  const int*   ei = (const int*)d_in[1];
  const int*   pt = (const int*)d_in[2];
  const float* W1 = (const float*)d_in[3];
  const float* b1 = (const float*)d_in[4];
  const float* W2 = (const float*)d_in[5];
  const float* b2 = (const float*)d_in[6];
  const float* Wc = (const float*)d_in[7];
  const float* bc = (const float*)d_in[8];
  float* out = (float*)d_out;

  const int Nn = in_sizes[0] / INDIM;   // 100000 (divisible by 16)
  const int E  = in_sizes[1] / 2;       // 3200000
  const int G  = in_sizes[2] - 1;       // 512
  const int* src = ei;
  const int* dst = ei + E;

  // workspace carve-out (~105 MB)
  char*  w = (char*)d_ws;
  size_t o = 0;
  auto take = [&](size_t bytes) -> char* {
    char* p = w + o; o += (bytes + 255) & ~(size_t)255; return p;
  };
  float*  dinv = (float*)take((size_t)Nn * 4);
  float*  bufA = (float*)take((size_t)Nn * HID * 4);
  float*  bufB = (float*)take((size_t)Nn * HID * 4);
  float*  hc   = (float*)take((size_t)Nn * 2 * 4);
  float*  acc3 = (float*)take((size_t)Nn * 2 * 4);
  __bf16* W1f  = (__bf16*)take((size_t)(INDIM / 32) * (HID / 16) * 512 * 2);
  __bf16* W2f  = (__bf16*)take((size_t)(HID / 32) * (HID / 16) * 512 * 2);
  (void)ws_size; (void)n_in; (void)out_size;

  auto cdiv = [](long a, long b) { return (int)((a + b - 1) / b); };
  auto smem = [](int K) { return (size_t)(16 * K + ((16 * K) >> 8) * 4) * 4; };
  const int T = 256;
  const int NF = Nn * HID;

  // --- normalization: deg init 1.0 (self-loop), count in-edges, rsqrt ---
  k_fill  <<<cdiv(Nn, T), T, 0, stream>>>(dinv, 1.0f, Nn);
  k_degree<<<cdiv(E,  T), T, 0, stream>>>(dst, dinv, E);
  k_rsqrt <<<cdiv(Nn, T), T, 0, stream>>>(dinv, Nn);

  // --- repack weights into bf16 WMMA fragments ---
  k_repack<<<cdiv((INDIM / 32) * (HID / 16) * 512, T), T, 0, stream>>>(W1, W1f, INDIM, HID / 16);
  k_repack<<<cdiv((HID   / 32) * (HID / 16) * 512, T), T, 0, stream>>>(W2, W2f, HID,   HID / 16);

  // --- layer 1: X@W1 -> bufA; aggregate -> bufB; finalize(relu) in bufB ---
  k_gemm_bf16<<<Nn / 16, T, smem(INDIM), stream>>>(x, W1f, bufA, INDIM, HID);
  k_fill     <<<cdiv(NF, T), T, 0, stream>>>(bufB, 0.0f, NF);
  k_scatter128<<<cdiv(E, 8), T, 0, stream>>>(bufA, src, dst, dinv, bufB, E);
  k_finalize <<<cdiv(NF, T), T, 0, stream>>>(bufA, dinv, b1, bufB, NF, HID, 1);

  // --- layer 2: h1@W2 -> bufA; aggregate -> bufB; finalize(relu) in bufB ---
  k_gemm_bf16<<<Nn / 16, T, smem(HID), stream>>>(bufB, W2f, bufA, HID, HID);
  k_fill     <<<cdiv(NF, T), T, 0, stream>>>(bufB, 0.0f, NF);
  k_scatter128<<<cdiv(E, 8), T, 0, stream>>>(bufA, src, dst, dinv, bufB, E);
  k_finalize <<<cdiv(NF, T), T, 0, stream>>>(bufA, dinv, b2, bufB, NF, HID, 1);

  // --- layer 3: h2@Wc -> hc; aggregate -> acc3; gather+finalize -> out ---
  k_linear_c<<<cdiv(Nn, T), T, 0, stream>>>(bufB, Wc, hc, Nn);
  k_fill    <<<cdiv(Nn * 2, T), T, 0, stream>>>(acc3, 0.0f, Nn * 2);
  k_scatter2<<<cdiv(E, T), T, 0, stream>>>(hc, src, dst, dinv, acc3, E);
  k_gather  <<<cdiv(G, T), T, 0, stream>>>(pt, hc, acc3, dinv, bc, out, G);
}
